// Qwen2MoeMLP_16587163697466
// MI455X (gfx1250) — matrix operand order repaired and checked by reading, not compiled, not run on hardware
//
#include <hip/hip_runtime.h>
#include <hip/hip_bf16.h>

typedef __attribute__((ext_vector_type(16))) _Float16 v16h;
typedef __attribute__((ext_vector_type(8)))  _Float16 v8h;
typedef __attribute__((ext_vector_type(8)))  float    v8f;
typedef __attribute__((ext_vector_type(4)))  int      v4i;

#define HID   2048
#define INTER 5632
#define TOK   32
#define QGRP  64   // quantization group size

union V16U { v16h v; v8h h8[2]; };

// Dequantize 16 contiguous int32 codes -> 16-bit B-fragment slice.
// w = c*s + bias with bias = -z*s hoisted per quant-group: one fma per code
// (compiler packs these into v_pk_fma_f32 + v_cvt_pk_f16_f32).
__device__ __forceinline__ v16h dequant16(const int* wp, float s, float bias) {
    union { v4i q[4]; int c[16]; } u;
    u.q[0] = *(const v4i*)(wp + 0);
    u.q[1] = *(const v4i*)(wp + 4);
    u.q[2] = *(const v4i*)(wp + 8);
    u.q[3] = *(const v4i*)(wp + 12);
    v16h b;
#pragma unroll
    for (int e = 0; e < 16; ++e)
        b[e] = (_Float16)__builtin_fmaf((float)u.c[e], s, bias);
    return b;
}

// ---------------------------------------------------------------------------
// Kernel 0: convert x (32x2048 f32) to f16 once, into workspace.
// ---------------------------------------------------------------------------
__global__ __launch_bounds__(256) void cvt_x_kernel(
    const float* __restrict__ x, _Float16* __restrict__ xh)
{
    const int i = blockIdx.x * 256 + threadIdx.x;   // grid covers TOK*HID
    xh[i] = (_Float16)x[i];
}

// ---------------------------------------------------------------------------
// Kernel 1: h[t, i] = silu(x @ Wg^T) * (x @ Wu^T), written as f16.
// One 16-col tile per block (352 blocks). 8 waves: waves 0-3 compute gate
// over K-quarters (512 each), waves 4-7 compute up. LDS reduce + SiLU.
// ---------------------------------------------------------------------------
__global__ __launch_bounds__(256) void mlp_gateup_kernel(
    const _Float16* __restrict__ xh,
    const int*   __restrict__ gWq, const float* __restrict__ gS, const float* __restrict__ gZ,
    const int*   __restrict__ uWq, const float* __restrict__ uS, const float* __restrict__ uZ,
    _Float16*    __restrict__ hout)
{
    __shared__ float gpart[4][TOK][16];
    __shared__ float upart[4][TOK][16];

    const int tid  = threadIdx.x;
    const int wave = tid >> 5;          // 0..7
    const int lane = tid & 31;
    const int isUp = wave >> 2;         // 0 = gate, 1 = up
    const int kw   = wave & 3;          // K-chunk id
    const int colBase = blockIdx.x * 16;

    const int*   Wq = isUp ? uWq : gWq;
    const float* Sc = isUp ? uS  : gS;
    const float* Zp = isUp ? uZ  : gZ;

    const int n = lane & 15;            // B column / C column
    const int g = lane >> 4;            // lane half-group
    const int row = colBase + n;        // weight row (inter index)
    const int* wrow = Wq + (size_t)row * HID;
    const float* srow = Sc + (size_t)row * (HID / QGRP);
    const float* zrow = Zp + (size_t)row * (HID / QGRP);
    const _Float16* r0 = xh + (size_t)n * HID;
    const _Float16* r1 = xh + (size_t)(n + 16) * HID;

    const int KW = HID / 4;             // 512, multiple of 32
    const int kStart = kw * KW;

    v8f c0 = {}, c1 = {};
    for (int k0 = kStart; k0 < kStart + KW; k0 += 32) {
        // A fragments: contiguous k = 16*g + s -> one 16-half run/lane (matches B)
        V16U a0, a1;
        a0.h8[0] = *(const v8h*)(r0 + k0 + 16 * g);
        a0.h8[1] = *(const v8h*)(r0 + k0 + 16 * g + 8);
        a1.h8[0] = *(const v8h*)(r1 + k0 + 16 * g);
        a1.h8[1] = *(const v8h*)(r1 + k0 + 16 * g + 8);
        // B fragment: 16 contiguous codes (one quant group covers them)
        const int kk = k0 + 16 * g;
        const int gi = kk / QGRP;
        const float s = srow[gi];
        const float bias = -zrow[gi] * s;
        v16h b = dequant16(wrow + kk, s, bias);

        c0 = __builtin_amdgcn_wmma_f32_16x16x32_f16(false, a0.v, false, b,
                                                    (short)0, c0, false, false);
        c1 = __builtin_amdgcn_wmma_f32_16x16x32_f16(false, a1.v, false, b,
                                                    (short)0, c1, false, false);
    }

    // C/D layout: VGPR r, lane l -> row = r + 8*(l>>4), col = l&15
    float (*part)[TOK][16] = isUp ? upart : gpart;
#pragma unroll
    for (int r = 0; r < 8; ++r) {
        part[kw][r + 8 * g][n]      = c0[r];
        part[kw][16 + r + 8 * g][n] = c1[r];
    }
    __syncthreads();

    // Reduce K-partials, SwiGLU with fast v_rcp, write f16 h to workspace
    for (int i = tid; i < TOK * 16; i += 256) {
        const int t = i >> 4, c = i & 15;
        const float gv = gpart[0][t][c] + gpart[1][t][c] +
                         gpart[2][t][c] + gpart[3][t][c];
        const float uv = upart[0][t][c] + upart[1][t][c] +
                         upart[2][t][c] + upart[3][t][c];
        const float sig = __builtin_amdgcn_rcpf(1.0f + __expf(-gv));
        const float hv = gv * sig * uv;
        hout[(size_t)t * INTER + colBase + c] = (_Float16)hv;
    }
}

// ---------------------------------------------------------------------------
// Kernel 2: out = h @ Wd^T. 16 output cols per block (128 blocks), split-K
// over 8 waves (704 = 22*32 each), LDS reduction, f32 store.
// ---------------------------------------------------------------------------
__global__ __launch_bounds__(256) void mlp_down_kernel(
    const _Float16* __restrict__ hin,
    const int*      __restrict__ dWq, const float* __restrict__ dS, const float* __restrict__ dZ,
    float*          __restrict__ out)
{
    __shared__ float part[8][TOK][16];

    const int tid  = threadIdx.x;
    const int wave = tid >> 5;          // 0..7
    const int lane = tid & 31;
    const int n = lane & 15;
    const int g = lane >> 4;
    const int colBase = blockIdx.x * 16;
    const int row = colBase + n;        // weight row (hid index)
    const int* wrow = dWq + (size_t)row * INTER;
    const float* srow = dS + (size_t)row * (INTER / QGRP);
    const float* zrow = dZ + (size_t)row * (INTER / QGRP);
    const _Float16* r0 = hin + (size_t)n * INTER;
    const _Float16* r1 = hin + (size_t)(n + 16) * INTER;

    const int KW = INTER / 8;           // 704, multiple of 32
    const int kStart = wave * KW;

    v8f c0 = {}, c1 = {};
    for (int k0 = kStart; k0 < kStart + KW; k0 += 32) {
        V16U a0, a1;
        a0.h8[0] = *(const v8h*)(r0 + k0 + 16 * g);
        a0.h8[1] = *(const v8h*)(r0 + k0 + 16 * g + 8);
        a1.h8[0] = *(const v8h*)(r1 + k0 + 16 * g);
        a1.h8[1] = *(const v8h*)(r1 + k0 + 16 * g + 8);

        const int kk = k0 + 16 * g;
        const int gi = kk / QGRP;
        const float s = srow[gi];
        const float bias = -zrow[gi] * s;
        v16h b = dequant16(wrow + kk, s, bias);

        c0 = __builtin_amdgcn_wmma_f32_16x16x32_f16(false, a0.v, false, b,
                                                    (short)0, c0, false, false);
        c1 = __builtin_amdgcn_wmma_f32_16x16x32_f16(false, a1.v, false, b,
                                                    (short)0, c1, false, false);
    }

#pragma unroll
    for (int r = 0; r < 8; ++r) {
        part[wave][r + 8 * g][n]      = c0[r];
        part[wave][16 + r + 8 * g][n] = c1[r];
    }
    __syncthreads();

    for (int i = tid; i < TOK * 16; i += 256) {
        const int t = i >> 4, c = i & 15;
        float s = 0.0f;
#pragma unroll
        for (int w = 0; w < 8; ++w) s += part[w][t][c];
        out[(size_t)t * HID + colBase + c] = s;
    }
}

extern "C" void kernel_launch(void* const* d_in, const int* in_sizes, int n_in,
                              void* d_out, int out_size, void* d_ws, size_t ws_size,
                              hipStream_t stream) {
    const float* x   = (const float*)d_in[0];
    const int*   gWq = (const int*)  d_in[1];
    const float* gS  = (const float*)d_in[2];
    const float* gZ  = (const float*)d_in[3];
    const int*   uWq = (const int*)  d_in[4];
    const float* uS  = (const float*)d_in[5];
    const float* uZ  = (const float*)d_in[6];
    const int*   dWq = (const int*)  d_in[7];
    const float* dS  = (const float*)d_in[8];
    const float* dZ  = (const float*)d_in[9];
    float* out = (float*)d_out;

    _Float16* xh = (_Float16*)d_ws;          // 32*2048 f16 = 128 KB
    _Float16* h  = xh + (size_t)TOK * HID;   // 32*5632 f16 = 352 KB

    cvt_x_kernel<<<(TOK * HID) / 256, 256, 0, stream>>>(x, xh);
    mlp_gateup_kernel<<<INTER / 16, 256, 0, stream>>>(xh, gWq, gS, gZ, uWq, uS, uZ, h);
    mlp_down_kernel<<<HID / 16, 256, 0, stream>>>(h, dWq, dS, dZ, out);
}
